// two_d_CXN_AMPS_19696720019794
// MI455X (gfx1250) — matrix operand order repaired and checked by reading, not compiled, not run on hardware
//
#include <hip/hip_runtime.h>

typedef unsigned short u16;
typedef unsigned int u32;
typedef __attribute__((ext_vector_type(16))) __bf16 v16bf;
typedef __attribute__((ext_vector_type(8))) float v8f;
typedef __attribute__((ext_vector_type(4))) float f32x4;

union BFrag {
    v16bf v;
    u16 u[16];
    uint4 q[2];
};

__device__ __forceinline__ u16 f2bf(float f) {
    u32 u = __float_as_uint(f);
    u += 0x7FFFu + ((u >> 16) & 1u);  // round-to-nearest-even bf16
    return (u16)(u >> 16);
}
__device__ __forceinline__ float bf2f(u16 h) {
    return __uint_as_float(((u32)h) << 16);
}

// ---------------------------------------------------------------------------
// Stage 1: Yt = (x @ W)^T in exact fp32, stored as split bf16 (hi + lo).
// Yt layout: (64, M) row-major so stage-2 B-fragment loads are contiguous.
// ---------------------------------------------------------------------------
__global__ void ltn_xw_t(const float* __restrict__ x, const float* __restrict__ W,
                         u16* __restrict__ Yh, u16* __restrict__ Yl, int M) {
    int t = blockIdx.x * blockDim.x + threadIdx.x;
    int n = t & 63;
    int m = t >> 6;
    if (m >= M) return;
    const float* xr = x + (size_t)m * 64;
    float v = 0.f;
#pragma unroll
    for (int c = 0; c < 64; ++c) v = fmaf(xr[c], W[c * 64 + n], v);
    u16 h = f2bf(v);
    size_t o = (size_t)n * M + m;
    Yh[o] = h;
    Yl[o] = f2bf(v - bf2f(h));
}

// ---------------------------------------------------------------------------
// Stage 2: K-sliced partial GEMM P = G1@Y1 [+ G2@Y2] over virtual-K slice,
// via v_wmma_f32_16x16x32_bf16 with in-register split-bf16 (hi+lo, 3-term).
// One wave owns a 16x64 output strip; block = 4 waves = 64 rows.
// G is a read-once HBM stream -> non-temporal loads + prefetch lookahead.
// ---------------------------------------------------------------------------
__device__ __forceinline__ v16bf load_frag(const u16* p) {
    BFrag f;
    f.q[0] = *(const uint4*)(p);
    f.q[1] = *(const uint4*)(p + 16);
    return f.v;
}

__device__ __forceinline__ void accum_seg(const float* __restrict__ G,
                                          const u16* __restrict__ Bh,
                                          const u16* __restrict__ Bl, int K,
                                          int k0, int k1,
                                          int mrow, int q, int n16, v8f acc[4]) {
    const float* grow = G + (size_t)mrow * K;
    for (int kk = k0; kk < k1; kk += 32) {
        __builtin_prefetch(grow + kk + 1024, 0, 1);  // global_prefetch lookahead
        // A fragment (16x32 bf16 layout): lane holds row M=lane%16,
        // K = g*16 + q*8 + e' for element groups g=0,1. Non-temporal stream.
        const float* gp = grow + kk + q * 8;
        f32x4 g0 = __builtin_nontemporal_load((const f32x4*)(gp));
        f32x4 g1 = __builtin_nontemporal_load((const f32x4*)(gp + 4));
        f32x4 g2 = __builtin_nontemporal_load((const f32x4*)(gp + 16));
        f32x4 g3 = __builtin_nontemporal_load((const f32x4*)(gp + 20));
        float av[16];
#pragma unroll
        for (int j = 0; j < 4; ++j) {
            av[j] = g0[j];
            av[4 + j] = g1[j];
            av[8 + j] = g2[j];
            av[12 + j] = g3[j];
        }
        BFrag ah, al;
#pragma unroll
        for (int e = 0; e < 16; ++e) {
            u16 h = f2bf(av[e]);
            ah.u[e] = h;
            al.u[e] = f2bf(av[e] - bf2f(h));
        }
#pragma unroll
        for (int t = 0; t < 4; ++t) {
            // B fragment (32x16): lane holds col N = t*16 + lane%16,
            // K = kk + g*16 + q*8 + e' for element groups g=0,1. Hot in L2/WGP$.
            size_t boff = (size_t)(t * 16 + n16) * K + kk + q * 8;
            v16bf bh = load_frag(Bh + boff);
            v16bf bl = load_frag(Bl + boff);
            acc[t] = __builtin_amdgcn_wmma_f32_16x16x32_bf16(
                false, ah.v, false, bh, (short)0, acc[t], false, false);
            acc[t] = __builtin_amdgcn_wmma_f32_16x16x32_bf16(
                false, ah.v, false, bl, (short)0, acc[t], false, false);
            acc[t] = __builtin_amdgcn_wmma_f32_16x16x32_bf16(
                false, al.v, false, bh, (short)0, acc[t], false, false);
        }
    }
}

__global__ __launch_bounds__(128) void gemm_slice_merge(
    const float* __restrict__ G1, const u16* __restrict__ B1h,
    const u16* __restrict__ B1l, int K1,
    const float* __restrict__ G2, const u16* __restrict__ B2h,
    const u16* __restrict__ B2l, int K2,
    float* __restrict__ P, int M) {
    int lane = threadIdx.x & 31;
    int wave = threadIdx.x >> 5;
    int q = lane >> 4;
    int n16 = lane & 15;
    int mbase = blockIdx.x * 64 + wave * 16;
    int mrow = mbase + n16;  // A-matrix row owned by this lane

    // Virtual-K slice handled by this gridDim.y index (all multiples of 32).
    int Ktot = K1 + K2;
    int ksz = Ktot / (int)gridDim.y;
    int k0 = blockIdx.y * ksz;
    int k1 = k0 + ksz;

    v8f acc[4] = {};
    int a0 = k0 < K1 ? k0 : K1;
    int a1 = k1 < K1 ? k1 : K1;
    if (a1 > a0) accum_seg(G1, B1h, B1l, K1, a0, a1, mrow, q, n16, acc);
    if (G2) {
        int b0 = k0 > K1 ? k0 - K1 : 0;
        int b1 = k1 > K1 ? k1 - K1 : 0;
        if (b1 > b0) accum_seg(G2, B2h, B2l, K2, b0, b1, mrow, q, n16, acc);
    }

    // D layout: VGPR r -> row M = q*8 + r, col N = lane%16. Write-once partial.
    float* Pm = P + (size_t)blockIdx.y * (size_t)M * 64;
#pragma unroll
    for (int t = 0; t < 4; ++t) {
#pragma unroll
        for (int r = 0; r < 8; ++r) {
            int row = mbase + q * 8 + r;
            __builtin_nontemporal_store(acc[t][r],
                                        Pm + (size_t)row * 64 + t * 16 + n16);
        }
    }
}

// Sum the 4 K-slice partials, apply ReLU, write final output.
__global__ void combine4_relu(const float* __restrict__ P, float* __restrict__ Out,
                              int n, size_t stride) {
    int i = blockIdx.x * blockDim.x + threadIdx.x;
    if (i >= n) return;
    float v = P[i] + P[stride + i] + P[2 * stride + i] + P[3 * stride + i];
    __builtin_nontemporal_store(v > 0.f ? v : 0.f, Out + i);
}

// ---------------------------------------------------------------------------
extern "C" void kernel_launch(void* const* d_in, const int* in_sizes, int n_in,
                              void* d_out, int out_size, void* d_ws, size_t ws_size,
                              hipStream_t stream) {
    (void)in_sizes; (void)n_in; (void)out_size; (void)ws_size;
    const float* xv   = (const float*)d_in[0];
    const float* xe   = (const float*)d_in[1];
    const float* xf   = (const float*)d_in[2];
    const float* Gv2v = (const float*)d_in[3];
    const float* Ge2v = (const float*)d_in[4];
    const float* Ge2e = (const float*)d_in[5];
    const float* Gf2e = (const float*)d_in[6];
    const float* Gf2f = (const float*)d_in[7];
    const float* Wvv  = (const float*)d_in[8];
    const float* Wev  = (const float*)d_in[9];
    const float* Wee  = (const float*)d_in[10];
    const float* Wfe  = (const float*)d_in[11];
    const float* Wff  = (const float*)d_in[12];
    float* out = (float*)d_out;

    const int NV = 8192, NE = 16384, NF = 8192;
    const int S = 4;  // K-slices per product (all slice bounds are %32 == 0)

    // Workspace: split-bf16 transposed Y matrices, then fp32 slice partials.
    u16* ws = (u16*)d_ws;
    u16* yv_h    = ws;                       // 64*NV
    u16* yv_l    = yv_h + (size_t)64 * NV;
    u16* ye_ev_h = yv_l + (size_t)64 * NV;   // 64*NE
    u16* ye_ev_l = ye_ev_h + (size_t)64 * NE;
    u16* ye_ee_h = ye_ev_l + (size_t)64 * NE;
    u16* ye_ee_l = ye_ee_h + (size_t)64 * NE;
    u16* yf_fe_h = ye_ee_l + (size_t)64 * NE;
    u16* yf_fe_l = yf_fe_h + (size_t)64 * NF;
    u16* yf_ff_h = yf_fe_l + (size_t)64 * NF;
    u16* yf_ff_l = yf_ff_h + (size_t)64 * NF;
    float* pv = (float*)(yf_ff_l + (size_t)64 * NF);  // S * NV*64
    float* pe = pv + (size_t)S * NV * 64;             // S * NE*64
    float* pf = pe + (size_t)S * NE * 64;             // S * NF*64

    // Stage 1: five small LTN pre-multiplies (exact fp32, then split).
    ltn_xw_t<<<(NV * 64) / 256, 256, 0, stream>>>(xv, Wvv, yv_h, yv_l, NV);
    ltn_xw_t<<<(NE * 64) / 256, 256, 0, stream>>>(xe, Wev, ye_ev_h, ye_ev_l, NE);
    ltn_xw_t<<<(NE * 64) / 256, 256, 0, stream>>>(xe, Wee, ye_ee_h, ye_ee_l, NE);
    ltn_xw_t<<<(NF * 64) / 256, 256, 0, stream>>>(xf, Wfe, yf_fe_h, yf_fe_l, NF);
    ltn_xw_t<<<(NF * 64) / 256, 256, 0, stream>>>(xf, Wff, yf_ff_h, yf_ff_l, NF);

    // Stage 2: K-sliced fused merges (WMMA split-bf16), partials per slice.
    gemm_slice_merge<<<dim3(NV / 64, S), 128, 0, stream>>>(
        Gv2v, yv_h, yv_l, NV, Ge2v, ye_ev_h, ye_ev_l, NE, pv, NV);
    gemm_slice_merge<<<dim3(NE / 64, S), 128, 0, stream>>>(
        Ge2e, ye_ee_h, ye_ee_l, NE, Gf2e, yf_fe_h, yf_fe_l, NF, pe, NE);
    gemm_slice_merge<<<dim3(NF / 64, S), 128, 0, stream>>>(
        Gf2f, yf_ff_h, yf_ff_l, NF, nullptr, nullptr, nullptr, 0, pf, NF);

    // Stage 3: reduce slices + ReLU.
    combine4_relu<<<(NV * 64) / 256, 256, 0, stream>>>(pv, out, NV * 64,
                                                       (size_t)NV * 64);
    combine4_relu<<<(NE * 64) / 256, 256, 0, stream>>>(pe, out + (size_t)NV * 64,
                                                       NE * 64, (size_t)NE * 64);
    combine4_relu<<<(NF * 64) / 256, 256, 0, stream>>>(
        pf, out + (size_t)NV * 64 + (size_t)NE * 64, NF * 64, (size_t)NF * 64);
}
